// GuidedAtt_10977936408673
// MI455X (gfx1250) — matrix operand order repaired and checked by reading, not compiled, not run on hardware
//
#include <hip/hip_runtime.h>

typedef __attribute__((ext_vector_type(8)))  float  v8f;
typedef __attribute__((ext_vector_type(16))) __bf16 v16bf;

typedef unsigned short u16;
typedef unsigned int   u32;
typedef unsigned long long u64;

static constexpr int BATCH   = 8;
static constexpr int NV      = 40962;
static constexpr int NF      = 81920;
static constexpr int TILE_V  = 128;

union BF16Frag {
  v16bf v;
  uint4 q[2];
};

__device__ __forceinline__ u16 f32_to_bf16(float f) {
  u32 u = __float_as_uint(f);
  u32 r = (u + 0x7FFFu + ((u >> 16) & 1u)) >> 16;
  return (u16)r;
}

// Branch-free tanh: prefer the CDNA5 HW transcendental (v_tanh_f32); else a
// straight-line exp2/rcp sequence (no saveexec slow path like libm tanhf).
__device__ __forceinline__ float fast_tanh(float x) {
#if __has_builtin(__builtin_amdgcn_tanhf)
  return __builtin_amdgcn_tanhf(x);
#elif __has_builtin(__builtin_amdgcn_tanh_f32)
  return __builtin_amdgcn_tanh_f32(x);
#else
  float ax2 = __builtin_fminf(__builtin_fabsf(x) * 2.8853900817779268f, 64.f);
  float e   = __builtin_amdgcn_exp2f(ax2);          // exp(2|x|)
  float r   = 1.f - 2.f * __builtin_amdgcn_rcpf(e + 1.f);
  return __builtin_copysignf(r, x);
#endif
}

__device__ __forceinline__ float fast_sigmoid(float x) {
  float e = __builtin_amdgcn_exp2f(__builtin_fmaxf(-x * 1.4426950408889634f, -126.f));
  return __builtin_amdgcn_rcpf(1.f + e);
}

// ---------------- Tensor Data Mover: contiguous global -> LDS block copy ----
// 2D descriptor, one row of `ndwords` 4-byte elements. Issued by one wave,
// tracked with TENSORcnt.
#if __has_builtin(__builtin_amdgcn_tensor_load_to_lds)
#define HAVE_TDM 1
typedef __attribute__((ext_vector_type(4))) unsigned int tdm_v4u;
typedef __attribute__((ext_vector_type(8))) int          tdm_v8i;
typedef __attribute__((ext_vector_type(4))) int          tdm_v4i;

__device__ __forceinline__ void tdm_copy_to_lds(u32 lds_off, const void* gptr,
                                                u32 ndwords) {
  u64 ga = (u64)(size_t)gptr;
  tdm_v4u g0;
  g0.x = 1u;                                          // count=1, user-mode D#
  g0.y = lds_off;                                     // lds_addr
  g0.z = (u32)(ga & 0xFFFFFFFFu);                     // global_addr[31:0]
  g0.w = (u32)((ga >> 32) & 0x01FFFFFFu) | (2u << 30);// global_addr[56:32]|type=2
  tdm_v8i g1;
  g1[0] = (int)(2u << 16);                            // data_size = 4 bytes
  g1[1] = (int)((ndwords & 0xFFFFu) << 16);           // tensor_dim0[15:0]
  g1[2] = (int)(((ndwords >> 16) & 0xFFFFu) | (1u << 16)); // dim0 hi | tensor_dim1=1
  g1[3] = (int)((ndwords & 0xFFFFu) << 16);           // tile_dim0
  g1[4] = 1;                                          // tile_dim1 = 1
  g1[5] = (int)ndwords;                               // tensor_dim0_stride
  g1[6] = 0;
  g1[7] = 0;
  tdm_v4i z4 = {0, 0, 0, 0};
#if defined(__clang_major__) && (__clang_major__ >= 23)
  tdm_v8i z8 = {0, 0, 0, 0, 0, 0, 0, 0};
  __builtin_amdgcn_tensor_load_to_lds(g0, g1, z4, z4, z8, 0);
#else
  __builtin_amdgcn_tensor_load_to_lds(g0, g1, z4, z4, 0);
#endif
  __builtin_amdgcn_s_wait_tensorcnt(0);
}
#else
#define HAVE_TDM 0
#endif

// ---------------------------------------------------------------- geometry
__global__ __launch_bounds__(256) void farea_kernel(
    const float* __restrict__ vnew, const int* __restrict__ f0,
    float* __restrict__ farea) {
  int i = blockIdx.x * blockDim.x + threadIdx.x;
  if (i >= BATCH * NF) return;
  int b = i / NF, f = i - b * NF;
  const float* vb = vnew + (size_t)b * NV * 3;
  int i0 = f0[f * 3 + 0], i1 = f0[f * 3 + 1], i2 = f0[f * 3 + 2];

  float ax = vb[i0*3+0], ay = vb[i0*3+1], az = vb[i0*3+2];
  float bx = vb[i1*3+0], by = vb[i1*3+1], bz = vb[i1*3+2];
  float cx = vb[i2*3+0], cy = vb[i2*3+1], cz = vb[i2*3+2];
  float ra = rsqrtf(ax*ax + ay*ay + az*az);
  float rb = rsqrtf(bx*bx + by*by + bz*bz);
  float rc = rsqrtf(cx*cx + cy*cy + cz*cz);
  ax *= ra; ay *= ra; az *= ra;
  bx *= rb; by *= rb; bz *= rb;
  cx *= rc; cy *= rc; cz *= rc;
  float e1x = ax - bx, e1y = ay - by, e1z = az - bz;
  float e2x = ax - cx, e2y = ay - cy, e2z = az - cz;
  float nx = e1y * e2z - e1z * e2y;
  float ny = e1z * e2x - e1x * e2z;
  float nz = e1x * e2y - e1y * e2x;
  farea[i] = sqrtf(nx*nx + ny*ny + nz*nz);
}

__global__ __launch_bounds__(256) void darea_kernel(
    const float* __restrict__ farea, const int* __restrict__ vfids,
    const float* __restrict__ area0, float* __restrict__ darea) {
  int i = blockIdx.x * blockDim.x + threadIdx.x;
  if (i >= BATCH * NV) return;
  int b = i / NV, v = i - b * NV;
  const float* fb = farea + (size_t)b * NF;
  float s = 0.f;
#pragma unroll
  for (int j = 0; j < 6; ++j) s += fb[vfids[v * 6 + j]];
  if (v < 12) s -= fb[0];
  darea[i] = s * (1.0f / 6.0f) - area0[v];
}

// ------------------------------------------------- weight prep (bf16 B-frags)
// Fragment order: [ktile][ntile][lane][16 elems]; lanes 0-15 -> col n=lane,
// K = e<8 ? e : 8+e; lanes 16-31 -> col n=lane-16, K = e<8 ? 8+e : 16+e  (CDNA5 16-bit B layout).
__global__ __launch_bounds__(256) void prep_weights(
    const float* __restrict__ W1, const float* __restrict__ Wp,
    u16* __restrict__ w1f, u16* __restrict__ wpf) {
  int tid = blockIdx.x * blockDim.x + threadIdx.x;
  const int total1 = 5 * 16 * 32 * 16;   // K padded to 160
  const int total2 = 4 * 16 * 32 * 16;   // K = 128
  if (tid < total1) {
    int e = tid & 15, lane = (tid >> 4) & 31, nt = (tid >> 9) & 15, kt = tid >> 13;
    int n = nt * 16 + (lane & 15);
    int k = kt * 32 + ((lane & 16) ? 8 : 0) + (e & 7) + ((e & 8) ? 16 : 0);
    float v = (k < 129) ? W1[k * 256 + n] : 0.0f;
    w1f[tid] = f32_to_bf16(v);
  } else if (tid < total1 + total2) {
    int t = tid - total1;
    int e = t & 15, lane = (t >> 4) & 31, nt = (t >> 9) & 15, kt = t >> 13;
    int n = nt * 16 + (lane & 15);
    int k = kt * 32 + ((lane & 16) ? 8 : 0) + (e & 7) + ((e & 8) ? 16 : 0);
    wpf[t] = f32_to_bf16(Wp[k * 256 + n]);
  }
}

// --------------------------------------------------------- attention (sa)
// smem: sW1 frags 80KB | sX 128x160 bf16 40KB | b1 1KB | W3 1KB  = 124928 B
__global__ __launch_bounds__(256) void attn_kernel(
    const float* __restrict__ dfeat, const float* __restrict__ darea,
    const u16* __restrict__ w1f, const float* __restrict__ b1,
    const float* __restrict__ W3, const float* __restrict__ b3,
    float* __restrict__ sa, int nTiles) {
  extern __shared__ char smem[];
  u16*   sW1 = (u16*)smem;            // 5*16*32*16 = 40960 u16 (LDS offset 0)
  u16*   sX  = sW1 + 40960;           // 128*160    = 20480 u16
  float* sB1 = (float*)(sX + 20480);  // 256
  float* sW3 = sB1 + 256;             // 256

  int tid = threadIdx.x;
  int b    = blockIdx.x / nTiles;
  int tile = blockIdx.x - b * nTiles;
  int v0   = tile * TILE_V;

  if (tid == 0) __builtin_prefetch(w1f, 0, 3);
#if HAVE_TDM
  if (tid < 32) tdm_copy_to_lds(0u, w1f, 5 * 16 * 32 * 16 / 2);  // 80KB block
#else
  {
    const uint4* src = (const uint4*)w1f;
    uint4* dst = (uint4*)sW1;
    for (int i = tid; i < 5120; i += 256) dst[i] = src[i];
  }
#endif
  if (tid < 256) { sB1[tid] = b1[tid]; sW3[tid] = W3[tid]; }

  const float* df_b = dfeat + (size_t)b * 128 * NV;
  const float* da_b = darea + (size_t)b * NV;
  for (int idx = tid; idx < TILE_V * 160; idx += 256) {
    int v = idx & 127, c = idx >> 7;
    int vg = v0 + v; if (vg >= NV) vg = NV - 1;
    float val;
    if (c < 128)       val = df_b[(size_t)c * NV + vg];
    else if (c == 128) val = da_b[vg];
    else               val = 0.f;
    sX[v * 160 + c] = f32_to_bf16(val);
  }
  __syncthreads();

  int lane = tid & 31, wave = tid >> 5;
  int row  = wave * 16 + (lane & 15);
  int koff = (lane & 16) ? 8 : 0;      // CDNA5 16-bit A layout half-select

  float p[8];
#pragma unroll
  for (int r = 0; r < 8; ++r) p[r] = 0.f;

  for (int nt = 0; nt < 16; ++nt) {
    v8f acc = {0.f, 0.f, 0.f, 0.f, 0.f, 0.f, 0.f, 0.f};
#pragma unroll
    for (int kt = 0; kt < 5; ++kt) {
      BF16Frag a, w;
      const uint4* ap = (const uint4*)(sX + row * 160 + kt * 32 + koff);
      a.q[0] = ap[0];          // K chunk .. +7
      a.q[1] = ap[2];          // K chunk +16 .. +23
      const uint4* wp = (const uint4*)(sW1 + (((kt * 16 + nt) * 32 + lane) << 4));
      w.q[0] = wp[0];
      w.q[1] = wp[1];
      acc = __builtin_amdgcn_wmma_f32_16x16x32_bf16(
          false, a.v, false, w.v, (short)0, acc, false, false);
    }
    int n = nt * 16 + (lane & 15);
    float bb = sB1[n], w3 = sW3[n];
#pragma unroll
    for (int r = 0; r < 8; ++r) p[r] += fast_tanh(acc[r] + bb) * w3;
  }

  float b3v = b3[0];
#pragma unroll
  for (int r = 0; r < 8; ++r) {
    float s = p[r];                         // lanes 0-15: M=row r; 16-31: M=8+r
    s += __shfl_xor(s, 1);
    s += __shfl_xor(s, 2);
    s += __shfl_xor(s, 4);
    s += __shfl_xor(s, 8);
    if ((lane & 15) == 0) {
      int vloc = wave * 16 + ((lane & 16) ? 8 : 0) + r;
      int vg = v0 + vloc;
      if (vg < NV)
        sa[(size_t)b * NV + vg] = fast_sigmoid(s + b3v);
    }
  }
}

// --------------------------------------------------------- projection (out)
// smem: sWp frags 64KB | sX 128x128 bf16 32KB | sSa 512B | sBp 1KB |
//       sO 64x132 f32 33KB  = 133632 B
__global__ __launch_bounds__(256) void proj_kernel(
    const float* __restrict__ dfeat, const float* __restrict__ sa,
    const u16* __restrict__ wpf, const float* __restrict__ bp,
    float* __restrict__ out, int nTiles) {
  extern __shared__ char smem[];
  u16*   sWp = (u16*)smem;            // 4*16*32*16 = 32768 u16 (LDS offset 0)
  u16*   sX  = sWp + 32768;           // 128*128    = 16384 u16
  float* sSa = (float*)(sX + 16384);  // 128
  float* sBp = sSa + 128;             // 256
  float* sO  = sBp + 256;             // 64 * 132

  int tid = threadIdx.x;
  int b    = blockIdx.x / nTiles;
  int tile = blockIdx.x - b * nTiles;
  int v0   = tile * TILE_V;

  if (tid == 0) __builtin_prefetch(wpf, 0, 3);
#if HAVE_TDM
  if (tid < 32) tdm_copy_to_lds(0u, wpf, 4 * 16 * 32 * 16 / 2);  // 64KB block
#else
  {
    const uint4* src = (const uint4*)wpf;
    uint4* dst = (uint4*)sWp;
    for (int i = tid; i < 4096; i += 256) dst[i] = src[i];
  }
#endif
  if (tid < 128) {
    int vg = v0 + tid; if (vg >= NV) vg = NV - 1;
    sSa[tid] = sa[(size_t)b * NV + vg];
  }
  if (tid < 256) sBp[tid] = bp[tid];
  __syncthreads();

  const float* df_b = dfeat + (size_t)b * 128 * NV;
  for (int idx = tid; idx < 128 * 128; idx += 256) {
    int v = idx & 127, c = idx >> 7;
    int vg = v0 + v; if (vg >= NV) vg = NV - 1;
    sX[v * 128 + c] = f32_to_bf16(df_b[(size_t)c * NV + vg] * sSa[v]);
  }
  __syncthreads();

  int lane = tid & 31, wave = tid >> 5;   // wave == M-tile (16 verts each)
  int row  = wave * 16 + (lane & 15);
  int koff = (lane & 16) ? 8 : 0;

  for (int chunk = 0; chunk < 4; ++chunk) {     // 4 x 64 output channels
#pragma unroll
    for (int ntl = 0; ntl < 4; ++ntl) {
      int nt = chunk * 4 + ntl;
      v8f acc = {0.f, 0.f, 0.f, 0.f, 0.f, 0.f, 0.f, 0.f};
#pragma unroll
      for (int kt = 0; kt < 4; ++kt) {
        BF16Frag a, w;
        const uint4* ap = (const uint4*)(sX + row * 128 + kt * 32 + koff);
        a.q[0] = ap[0];
        a.q[1] = ap[2];
        const uint4* wp = (const uint4*)(sWp + (((kt * 16 + nt) * 32 + lane) << 4));
        w.q[0] = wp[0];
        w.q[1] = wp[1];
        acc = __builtin_amdgcn_wmma_f32_16x16x32_bf16(
            false, a.v, false, w.v, (short)0, acc, false, false);
      }
      int n = nt * 16 + (lane & 15);
      float bb = sBp[n];
#pragma unroll
      for (int r = 0; r < 8; ++r) {
        float val = acc[r] + bb;
        val = val > 0.f ? val : 0.f;
        int vloc = wave * 16 + ((lane & 16) ? 8 : 0) + r;
        sO[(ntl * 16 + (lane & 15)) * 132 + vloc] = val;   // padded: no bank clash
      }
    }
    __syncthreads();
    // coalesced channel-major store: 64 rows x 128 contiguous verts
    for (int idx = tid; idx < 64 * 128; idx += 256) {
      int v = idx & 127, cl = idx >> 7;
      int cg = chunk * 64 + cl;
      int vg = v0 + v;
      if (vg < NV)
        out[((size_t)b * 256 + cg) * NV + vg] = sO[cl * 132 + v];
    }
    __syncthreads();
  }
}

// ---------------------------------------------------------------- launcher
extern "C" void kernel_launch(void* const* d_in, const int* in_sizes, int n_in,
                              void* d_out, int out_size, void* d_ws, size_t ws_size,
                              hipStream_t stream) {
  (void)in_sizes; (void)n_in; (void)out_size; (void)ws_size;
  const float* vnew  = (const float*)d_in[0];
  const float* dfeat = (const float*)d_in[1];
  const int*   f0    = (const int*)d_in[2];
  const int*   vfids = (const int*)d_in[3];
  const float* area0 = (const float*)d_in[4];
  const float* W1    = (const float*)d_in[5];
  const float* b1    = (const float*)d_in[6];
  const float* W3    = (const float*)d_in[7];
  const float* b3    = (const float*)d_in[8];
  const float* Wp    = (const float*)d_in[9];
  const float* bp    = (const float*)d_in[10];
  float* out = (float*)d_out;

  char* ws = (char*)d_ws;
  const size_t szFarea = (size_t)BATCH * NF * sizeof(float);   // 2,621,440
  const size_t szBV    = (size_t)BATCH * NV * sizeof(float);   // 1,310,784
  float* farea = (float*)(ws);
  float* darea = (float*)(ws + szFarea);
  float* saBuf = (float*)(ws + szFarea + szBV);
  u16*   w1f   = (u16*)(ws + szFarea + 2 * szBV);
  u16*   wpf   = w1f + 5 * 16 * 32 * 16;

  farea_kernel<<<(BATCH * NF + 255) / 256, 256, 0, stream>>>(vnew, f0, farea);
  darea_kernel<<<(BATCH * NV + 255) / 256, 256, 0, stream>>>(farea, vfids, area0, darea);
  prep_weights<<<(5 * 16 * 32 * 16 + 4 * 16 * 32 * 16 + 255) / 256, 256, 0, stream>>>(
      W1, Wp, w1f, wpf);

  int nTiles = (NV + TILE_V - 1) / TILE_V;   // 321
  size_t attnSmem = (40960 + 20480) * sizeof(u16) + (256 + 256) * sizeof(float);
  size_t projSmem = (32768 + 16384) * sizeof(u16) + (128 + 256 + 64 * 132) * sizeof(float);

  attn_kernel<<<BATCH * nTiles, 256, attnSmem, stream>>>(
      dfeat, darea, w1f, b1, W3, b3, saBuf, nTiles);
  proj_kernel<<<BATCH * nTiles, 256, projSmem, stream>>>(
      dfeat, saBuf, wpf, bp, out, nTiles);
}
